// MSSSIML1_Loss_65283502899205
// MI455X (gfx1250) — matrix operand order repaired and checked by reading, not compiled, not run on hardware
//
// MS-SSIM-L1 loss for MI455X (gfx1250, wave32, WMMA).
//
// Roofline: inputs 128MB, whole working set fits the 192MB L2 -> bandwidth
// bound; FLOPs (~10G with separable convs) are negligible vs WMMA peak.
// Design: full fusion (no conv maps materialized), separable Gaussian convs
// expressed as 16x16x32 f16 WMMAs against banded Toeplitz weight matrices,
// f32 accumulation, f64 atomic reductions for the scalar statistics.
// Cross-lane H->V layout fixup uses v_permlanex16_b32 (VALU, co-executes with
// the XDL WMMA pipe) on packed f16 pairs instead of ds_bpermute.

#include <hip/hip_runtime.h>

typedef __attribute__((ext_vector_type(16))) _Float16 v16h;
typedef __attribute__((ext_vector_type(8)))  _Float16 v8h;
typedef __attribute__((ext_vector_type(4)))  _Float16 v4h;
typedef __attribute__((ext_vector_type(2)))  _Float16 v2h;
typedef __attribute__((ext_vector_type(8)))  float    v8f;
typedef __attribute__((ext_vector_type(4)))  float    v4f;
typedef __attribute__((ext_vector_type(8)))  int      v8i;

#define MSA 0.84f

// 11-tap gaussian, sigma=1.5, normalized (matches _create_window(11,1.5)).
__device__ __constant__ float W11[11] = {
    0.00102838f, 0.00759875f, 0.03600077f, 0.10936069f, 0.21300553f,
    0.26601172f, 0.21300553f, 0.10936069f, 0.03600077f, 0.00759875f,
    0.00102838f};

__device__ __constant__ float MSW[5] = {0.0448f, 0.2856f, 0.3001f, 0.2363f, 0.1333f};

// ---------------- WMMA helpers ----------------

static __device__ inline v8f wmma16(v16h a, v16h b, v8f c) {
  // D = A(16x32 f16) x B(32x16 f16) + C(16x16 f32)
  return __builtin_amdgcn_wmma_f32_16x16x32_f16(false, a, false, b,
                                                (short)0, c, false, false);
}

// lane i <-> lane i^16 swap as a VALU op (identity nibble selects).
static __device__ inline int xswap16_i(int v) {
  return __builtin_amdgcn_permlanex16(v, v, 0x76543210u, 0xfedcba98u, false,
                                      false);
}
static __device__ inline float xswap16_f(float x) {
  return __int_as_float(xswap16_i(__float_as_int(x)));
}

// A-image fragment (16x32 f16): lane holds row (lane&15)+row0; K chunks
// {kb..kb+7, kb+16..kb+23}, kb = lane<16 ? 0 : 8  (per ISA 7.12.2).
static __device__ inline v16h afrag(const _Float16* t, int pitch, int row0,
                                    int col0, int lane) {
  int r  = row0 + (lane & 15);
  int kb = (lane < 16) ? 0 : 8;
  const _Float16* p = t + r * pitch + col0 + kb;
  v8h lo = *(const v8h*)(p);
  v8h hi = *(const v8h*)(p + 16);
  v16h o;
#pragma unroll
  for (int j = 0; j < 8; ++j) { o[j] = lo[j]; o[8 + j] = hi[j]; }
  return o;
}

// B Toeplitz (32x16): B[k][n] = w[koff + k - n] (horizontal conv weights).
static __device__ inline v16h btoe(const float* w, int nw, int koff, int lane) {
  int n  = lane & 15;
  int kb = ((lane < 16) ? 0 : 8) + koff;
  v16h o;
#pragma unroll
  for (int j = 0; j < 16; ++j) {
    int t = ((j < 8) ? (kb + j) : (kb + 16 + (j - 8))) - n;
    o[j] = (t >= 0 && t < nw) ? (_Float16)w[t] : (_Float16)0.0f;
  }
  return o;
}

// A Toeplitz (16x32): A[m][k] = w[koff + k - m] (vertical conv weights).
static __device__ inline v16h atoe(const float* w, int nw, int koff, int lane) {
  int m  = lane & 15;
  int kb = (lane < 16) ? 0 : 16;
  v16h o;
#pragma unroll
  for (int j = 0; j < 16; ++j) {
    int k = kb + j + koff;
    int t = k - m;
    o[j] = (t >= 0 && t < nw) ? (_Float16)w[t] : (_Float16)0.0f;
  }
  return o;
}

// Convert two stacked H-pass D fragments (rows base..base+15 and
// base+16..base+31) into a V-pass B fragment (32x16 f16). Pack f32 pairs to
// f16 first (v_cvt_pk_f16_f32), then swap lane halves with permlanex16.
static __device__ inline v16h bfrag(v8f dlo, v8f dhi, int lane) {
  int plo[4], phi[4], slo[4], shi[4];
#pragma unroll
  for (int i = 0; i < 4; ++i) {
    v2h a; a[0] = (_Float16)dlo[2 * i]; a[1] = (_Float16)dlo[2 * i + 1];
    v2h b; b[0] = (_Float16)dhi[2 * i]; b[1] = (_Float16)dhi[2 * i + 1];
    plo[i] = __builtin_bit_cast(int, a);
    phi[i] = __builtin_bit_cast(int, b);
    slo[i] = xswap16_i(plo[i]);
    shi[i] = xswap16_i(phi[i]);
  }
  bool low = lane < 16;
  v8i w;
#pragma unroll
  for (int i = 0; i < 4; ++i) {
    w[i]     = low ? plo[i] : shi[i];  // lanes 0-15: rows base+0..7; lanes 16-31: rows base+16..23
    w[4 + i] = low ? slo[i] : phi[i];  // lanes 0-15: rows base+8..15; lanes 16-31: rows base+24..31
  }
  return __builtin_bit_cast(v16h, w);
}

static __device__ inline float wave_sum(float x) {
  x += xswap16_f(x);
  for (int off = 8; off > 0; off >>= 1) x += __shfl_xor(x, off, 32);
  return x;
}

// ---------------- order-preserving float<->uint for atomic min/max ----------

static __device__ inline unsigned enc_ord(float f) {
  unsigned u = __float_as_uint(f);
  return (u & 0x80000000u) ? ~u : (u | 0x80000000u);
}
static __device__ inline float dec_ord(unsigned k) {
  unsigned u = (k & 0x80000000u) ? (k & 0x7fffffffu) : ~k;
  return __uint_as_float(u);
}

// ---------------- kernels ----------------

__global__ __launch_bounds__(256) void init_kernel(double* accs, unsigned* mm) {
  int t = threadIdx.x;
  if (t < 161) accs[t] = 0.0;  // ssim[80] + cs[80] + l1[1]
  if (t < 5) {
    mm[2 * t]     = enc_ord(-__builtin_inff());  // running max identity
    mm[2 * t + 1] = enc_ord(__builtin_inff());   // running min identity
  }
}

__global__ __launch_bounds__(256) void minmax_kernel(const float* __restrict__ x,
                                                     size_t n, unsigned* mm) {
  __shared__ float smax[256], smin[256];
  int tid = threadIdx.x;
  float mx = -__builtin_inff(), mn = __builtin_inff();
  for (size_t i = (size_t)blockIdx.x * 256 + tid; i < n;
       i += (size_t)gridDim.x * 256) {
    float v = x[i];
    mx = fmaxf(mx, v);
    mn = fminf(mn, v);
  }
  smax[tid] = mx; smin[tid] = mn;
  __syncthreads();
  for (int s = 128; s > 0; s >>= 1) {
    if (tid < s) {
      smax[tid] = fmaxf(smax[tid], smax[tid + s]);
      smin[tid] = fminf(smin[tid], smin[tid + s]);
    }
    __syncthreads();
  }
  if (tid == 0) {
    atomicMax(&mm[0], enc_ord(smax[0]));
    atomicMin(&mm[1], enc_ord(smin[0]));
  }
}

// 2x2 average pool; for img1 also reduces next level's min/max.
__global__ __launch_bounds__(256) void pool_kernel(const float* __restrict__ in,
                                                   float* __restrict__ out,
                                                   int inW, int outH, int outW,
                                                   int B, unsigned* mm_next) {
  size_t n   = (size_t)B * outH * outW;
  size_t idx = (size_t)blockIdx.x * 256 + threadIdx.x;
  float v = 0.0f;
  bool valid = idx < n;
  if (valid) {
    int    x = (int)(idx % outW);
    size_t t = idx / outW;
    int    y = (int)(t % outH);
    int    b = (int)(t / outH);
    const float* p = in + (size_t)b * (2 * outH) * inW;
    const float* r0 = p + (size_t)(2 * y) * inW + 2 * x;
    const float* r1 = r0 + inW;
    v = 0.25f * (r0[0] + r0[1] + r1[0] + r1[1]);
    out[idx] = v;
  }
  if (mm_next) {
    __shared__ float smax[256], smin[256];
    int tid = threadIdx.x;
    smax[tid] = valid ? v : -__builtin_inff();
    smin[tid] = valid ? v : __builtin_inff();
    __syncthreads();
    for (int s = 128; s > 0; s >>= 1) {
      if (tid < s) {
        smax[tid] = fmaxf(smax[tid], smax[tid + s]);
        smin[tid] = fminf(smin[tid], smin[tid + s]);
      }
      __syncthreads();
    }
    if (tid == 0) {
      atomicMax(&mm_next[0], enc_ord(smax[0]));
      atomicMin(&mm_next[1], enc_ord(smin[0]));
    }
  }
}

// One wave computes one 16x16 output tile of the SSIM level: 5 separable
// 11x11 Gaussian convs as 15 v_wmma_f32_16x16x32_f16, then SSIM/CS math and
// a wave-reduced f64 atomic accumulation of the per-batch sums.
__global__ __launch_bounds__(32) void ssim_kernel(
    const float* __restrict__ img1, const float* __restrict__ img2, int H,
    int W, int outH, int outW, const unsigned* __restrict__ mm,
    double* __restrict__ ssim_acc, double* __restrict__ cs_acc) {
  __shared__ __align__(16) _Float16 t1[32 * 32];
  __shared__ __align__(16) _Float16 t2[32 * 32];
  int lane = threadIdx.x;
  int b    = blockIdx.z;
  int oy0  = blockIdx.y * 16, ox0 = blockIdx.x * 16;
  const float* p1 = img1 + (size_t)b * H * W;
  const float* p2 = img2 + (size_t)b * H * W;

  // stage 32x32 f16 tiles: float4 loads (aligned: ox0%16==0, W%4==0),
  // packed 8-byte LDS stores.
  for (int it = 0; it < 8; ++it) {
    int r  = it * 4 + (lane >> 3);
    int c  = (lane & 7) * 4;
    int gy = oy0 + r, gx = ox0 + c;
    v4f a = {}, bb = {};
    if (gy < H) {
      const float* q1 = p1 + (size_t)gy * W;
      const float* q2 = p2 + (size_t)gy * W;
      if (gx + 3 < W) {
        a  = *(const v4f*)(q1 + gx);
        bb = *(const v4f*)(q2 + gx);
      } else {
#pragma unroll
        for (int j = 0; j < 4; ++j)
          if (gx + j < W) { a[j] = q1[gx + j]; bb[j] = q2[gx + j]; }
      }
    }
    v4h ah, bh;
#pragma unroll
    for (int j = 0; j < 4; ++j) {
      ah[j] = (_Float16)a[j];
      bh[j] = (_Float16)bb[j];
    }
    *(v4h*)(t1 + r * 32 + c) = ah;
    *(v4h*)(t2 + r * 32 + c) = bh;
  }
  __syncthreads();

  v8f  zero = {};
  v16h Bt = btoe(W11, 11, 0, lane);
  v16h At = atoe(W11, 11, 0, lane);

  v16h x1a = afrag(t1, 32, 0, 0, lane), x1b = afrag(t1, 32, 16, 0, lane);
  v16h x2a = afrag(t2, 32, 0, 0, lane), x2b = afrag(t2, 32, 16, 0, lane);

  v8f mu1, mu2, s11, s22, s12;
  { v8f d0 = wmma16(x1a, Bt, zero), d1 = wmma16(x1b, Bt, zero);
    mu1 = wmma16(At, bfrag(d0, d1, lane), zero); }
  { v8f d0 = wmma16(x2a, Bt, zero), d1 = wmma16(x2b, Bt, zero);
    mu2 = wmma16(At, bfrag(d0, d1, lane), zero); }
  { v8f d0 = wmma16(x1a * x1a, Bt, zero), d1 = wmma16(x1b * x1b, Bt, zero);
    s11 = wmma16(At, bfrag(d0, d1, lane), zero); }
  { v8f d0 = wmma16(x2a * x2a, Bt, zero), d1 = wmma16(x2b * x2b, Bt, zero);
    s22 = wmma16(At, bfrag(d0, d1, lane), zero); }
  { v8f d0 = wmma16(x1a * x2a, Bt, zero), d1 = wmma16(x1b * x2b, Bt, zero);
    s12 = wmma16(At, bfrag(d0, d1, lane), zero); }

  // data-dependent value range (reference heuristic)
  float fmax = dec_ord(mm[0]), fmin = dec_ord(mm[1]);
  float L  = (fmax > 128.0f ? 255.0f : 1.0f) - (fmin < -0.5f ? -1.0f : 0.0f);
  float C1 = (0.01f * L) * (0.01f * L);
  float C2 = (0.03f * L) * (0.03f * L);

  float ss = 0.0f, cc = 0.0f;
#pragma unroll
  for (int i = 0; i < 8; ++i) {
    int row = (lane < 16) ? i : (8 + i);
    if ((oy0 + row) < outH && (ox0 + (lane & 15)) < outW) {
      float m1 = mu1[i], m2 = mu2[i];
      float m1s = m1 * m1, m2s = m2 * m2, m12 = m1 * m2;
      float g1 = s11[i] - m1s, g2 = s22[i] - m2s, g12 = s12[i] - m12;
      float v1 = 2.0f * g12 + C2, v2 = g1 + g2 + C2;
      cc += v1 / v2;
      ss += ((2.0f * m12 + C1) * v1) / ((m1s + m2s + C1) * v2);
    }
  }
  ss = wave_sum(ss);
  cc = wave_sum(cc);
  if (lane == 0) {
    atomicAdd(&ssim_acc[b], (double)ss);
    atomicAdd(&cs_acc[b], (double)cc);
  }
}

// 33x33 separable Gaussian conv of |img1-img2| (valid), summed.
// K span is 48 -> two chained K=32 Toeplitz segments via the C accumulator:
// 6 horizontal + 2 vertical WMMAs per 16x16 tile.
__global__ __launch_bounds__(32) void l1_kernel(const float* __restrict__ img1,
                                                const float* __restrict__ img2,
                                                int H, int W, int outH, int outW,
                                                double* __restrict__ l1_acc) {
  __shared__ __align__(16) _Float16 td[48 * 64];
  __shared__ float w33[33];
  int lane = threadIdx.x;
  if (lane == 0) {  // gaussian(33, sigma=0.8), normalized
    float tmp[33], s = 0.0f;
    for (int i = 0; i < 33; ++i) {
      float d = (float)(i - 16);
      float g = expf(-d * d / (2.0f * 0.8f * 0.8f));
      tmp[i] = g; s += g;
    }
    for (int i = 0; i < 33; ++i) w33[i] = tmp[i] / s;
  }
  int oy0 = blockIdx.y * 16, ox0 = blockIdx.x * 16;
  int b   = blockIdx.z;
  const float* p1 = img1 + (size_t)b * H * W;
  const float* p2 = img2 + (size_t)b * H * W;

  // stage 48x64 f16 |diff| tile: float4 loads, packed 8-byte LDS stores.
  for (int it = 0; it < 24; ++it) {
    int r  = it * 2 + (lane >> 4);
    int c  = (lane & 15) * 4;
    int gy = oy0 + r, gx = ox0 + c;
    v4f d = {};
    if (gy < H) {
      const float* q1 = p1 + (size_t)gy * W;
      const float* q2 = p2 + (size_t)gy * W;
      if (gx + 3 < W) {
        v4f a = *(const v4f*)(q1 + gx);
        v4f bb = *(const v4f*)(q2 + gx);
#pragma unroll
        for (int j = 0; j < 4; ++j) d[j] = fabsf(a[j] - bb[j]);
      } else {
#pragma unroll
        for (int j = 0; j < 4; ++j)
          if (gx + j < W) d[j] = fabsf(q1[gx + j] - q2[gx + j]);
      }
    }
    v4h dh;
#pragma unroll
    for (int j = 0; j < 4; ++j) dh[j] = (_Float16)d[j];
    *(v4h*)(td + r * 64 + c) = dh;
  }
  __syncthreads();

  v8f  zero = {};
  v16h Bt0 = btoe(w33, 33, 0, lane), Bt1 = btoe(w33, 33, 32, lane);
  v16h At0 = atoe(w33, 33, 0, lane), At1 = atoe(w33, 33, 32, lane);

  v8f h0, h1, h2;
  { v16h aL = afrag(td, 64, 0, 0, lane),  aR = afrag(td, 64, 0, 32, lane);
    h0 = wmma16(aR, Bt1, wmma16(aL, Bt0, zero)); }
  { v16h aL = afrag(td, 64, 16, 0, lane), aR = afrag(td, 64, 16, 32, lane);
    h1 = wmma16(aR, Bt1, wmma16(aL, Bt0, zero)); }
  { v16h aL = afrag(td, 64, 32, 0, lane), aR = afrag(td, 64, 32, 32, lane);
    h2 = wmma16(aR, Bt1, wmma16(aL, Bt0, zero)); }

  v16h b0 = bfrag(h0, h1, lane);     // rows 0..31
  v16h b1 = bfrag(h2, zero, lane);   // rows 32..47 (48..63 are zero)
  v8f  d  = wmma16(At1, b1, wmma16(At0, b0, zero));

  float s = 0.0f;
#pragma unroll
  for (int i = 0; i < 8; ++i) {
    int row = (lane < 16) ? i : (8 + i);
    if ((oy0 + row) < outH && (ox0 + (lane & 15)) < outW) s += d[i];
  }
  s = wave_sum(s);
  if (lane == 0) atomicAdd(l1_acc, (double)s);
}

// Reproduces the reference broadcast: prod over [4,B,5] of
// mcs[l,b]^w[j] * ssims[4,b]^w[j], then loss combine.
__global__ void final_kernel(const double* __restrict__ ssim_acc,
                             const double* __restrict__ cs_acc,
                             const double* __restrict__ l1_acc,
                             float* __restrict__ out) {
  if (threadIdx.x != 0 || blockIdx.x != 0) return;
  const int od[5] = {1014, 502, 246, 118, 54};
  float sm[5][16], cm[5][16];
  for (int l = 0; l < 5; ++l) {
    double cnt = (double)od[l] * (double)od[l];
    for (int b = 0; b < 16; ++b) {
      sm[l][b] = (float)(ssim_acc[l * 16 + b] / cnt);
      cm[l][b] = (float)(cs_acc[l * 16 + b] / cnt);
    }
  }
  double prod = 1.0;
  for (int l = 0; l < 4; ++l)
    for (int b = 0; b < 16; ++b)
      for (int j = 0; j < 5; ++j)
        prod *= (double)(powf(cm[l][b], MSW[j]) * powf(sm[4][b], MSW[j]));
  float ms  = (float)prod;
  float l1m = (float)(l1_acc[0] / (16.0 * 992.0 * 992.0));
  out[0] = MSA * (1.0f - ms) + (1.0f - MSA) * l1m;
}

// ---------------- host launch ----------------

extern "C" void kernel_launch(void* const* d_in, const int* in_sizes, int n_in,
                              void* d_out, int out_size, void* d_ws,
                              size_t ws_size, hipStream_t stream) {
  (void)in_sizes; (void)n_in; (void)out_size; (void)ws_size;
  const int B = 16, H0 = 1024;
  const int dims[5] = {1024, 512, 256, 128, 64};

  const float* img1 = (const float*)d_in[0];
  const float* img2 = (const float*)d_in[1];
  float* out = (float*)d_out;

  char*     ws       = (char*)d_ws;
  double*   ssim_acc = (double*)ws;            // [5][16]
  double*   cs_acc   = ssim_acc + 80;          // [5][16]
  double*   l1_acc   = cs_acc + 80;            // [1]
  unsigned* mm       = (unsigned*)(l1_acc + 1);// [5][2] {max,min} keys

  // pooled pyramids (levels 1..4), ~42.5 MB total in ws
  float* p = (float*)(ws + 4096);
  float* m1[5] = {nullptr}; float* m2[5] = {nullptr};
  size_t off = 0;
  for (int l = 1; l < 5; ++l) { m1[l] = p + off; off += (size_t)B * dims[l] * dims[l]; }
  for (int l = 1; l < 5; ++l) { m2[l] = p + off; off += (size_t)B * dims[l] * dims[l]; }
  const float* s1[5] = {img1, m1[1], m1[2], m1[3], m1[4]};
  const float* s2[5] = {img2, m2[1], m2[2], m2[3], m2[4]};

  init_kernel<<<1, 256, 0, stream>>>(ssim_acc, mm);
  minmax_kernel<<<1024, 256, 0, stream>>>(img1, (size_t)B * H0 * H0, mm);

  for (int l = 0; l < 5; ++l) {
    int Hl = dims[l], outl = Hl - 10;
    int tiles = (outl + 15) / 16;
    ssim_kernel<<<dim3(tiles, tiles, B), 32, 0, stream>>>(
        s1[l], s2[l], Hl, Hl, outl, outl, mm + 2 * l, ssim_acc + 16 * l,
        cs_acc + 16 * l);
    if (l < 4) {
      int oh = dims[l + 1];
      size_t n = (size_t)B * oh * oh;
      int blocks = (int)((n + 255) / 256);
      pool_kernel<<<blocks, 256, 0, stream>>>(s1[l], m1[l + 1], Hl, oh, oh, B,
                                              mm + 2 * (l + 1));
      pool_kernel<<<blocks, 256, 0, stream>>>(s2[l], m2[l + 1], Hl, oh, oh, B,
                                              nullptr);
    }
  }

  {  // 33x33 L1 conv, out 992x992
    int outl = H0 - 32;
    int tiles = (outl + 15) / 16;  // 62
    l1_kernel<<<dim3(tiles, tiles, B), 32, 0, stream>>>(img1, img2, H0, H0,
                                                        outl, outl, l1_acc);
  }

  final_kernel<<<1, 1, 0, stream>>>(ssim_acc, cs_acc, l1_acc, out);
}
